// SASRec_39067022525076
// MI455X (gfx1250) — matrix operand order repaired and checked by reading, not compiled, not run on hardware
//
#include <hip/hip_runtime.h>

// ---------------------------------------------------------------------------
// SASRec forward for MI455X (gfx1250, wave32, WMMA).
// All GEMM-shaped math -> v_wmma_f32_16x16x32_bf16 (bf16 in, f32 accum).
// Attention: K staged bf16 in LDS, V staged via global_load_async_to_lds_b128
// (ASYNCcnt), score rows in LDS f32, softmax in f32 VALU, P@V through WMMA.
// ---------------------------------------------------------------------------

#define B_BATCH 512
#define L_SEQ   200
#define D_MODEL 64
#define NH_C    2
#define E_EXP   4
#define TOPK_C  2
#define NB_C    2
#define ALPHA_C 0.5f
#define TAU_C   1.0f
#define LP      224        // L padded to 14 tiles of 16 (7 k-tiles of 32)
#define R_TOT   (B_BATCH * L_SEQ)

typedef __attribute__((ext_vector_type(16))) __bf16 v16bf;
typedef __attribute__((ext_vector_type(8)))  float  v8f;

// --- WMMA fragment index helpers (wave32; A/B k order contiguous, not ISA table order) --------
// A (16x32, 16-bit): lane -> M = lane&15, half = lane>>4.
// VGPR i in [0,7]: K pair = half*16 + 2i (same k order as B).
__device__ __forceinline__ int afrag_k(int i, int half) {
  return half * 16 + 2 * i;
}
// B (32x16, 16-bit): lane -> N = lane&15 ; VGPR i holds K = half*16 + 2i, +1.
// C/D (16x16, f32): lane -> N = lane&15 ; VGPR r -> M = r + 8*half.

__device__ __forceinline__ v8f wmma_bf16(const v16bf& a, const v16bf& b, const v8f& c) {
  return __builtin_amdgcn_wmma_f32_16x16x32_bf16(false, a, false, b, (short)0, c,
                                                 false, false);
}

// Wave-relative LDS byte offset of a __shared__ object (addrspacecast to AS3).
__device__ __forceinline__ unsigned lds_off(void* p) {
  return (unsigned)(unsigned long long)(__attribute__((address_space(3))) char*)p;
}

// ---------------------------------------------------------------------------
// Embedding gather: x = item_emb[seq]*sqrt(D) + pos_emb[(l+1)*(seq!=0)]
// ---------------------------------------------------------------------------
__global__ __launch_bounds__(256) void embed_kernel(
    const int* __restrict__ log_seqs, const float* __restrict__ item_emb,
    const float* __restrict__ pos_emb, float* __restrict__ X) {
  int idx = blockIdx.x * 256 + threadIdx.x;
  if (idx >= R_TOT * D_MODEL) return;
  int r = idx >> 6, d = idx & 63;
  int l = r % L_SEQ;
  int tok = log_seqs[r];
  float v = item_emb[(size_t)tok * D_MODEL + d] * 8.0f;  // sqrt(64)
  int pos = (tok != 0) ? (l + 1) : 0;
  v += pos_emb[(size_t)pos * D_MODEL + d];
  X[idx] = v;
}

// ---------------------------------------------------------------------------
// Row LayerNorm over D=64 (optionally Y = LN(X+Add)), wave32 shuffle reduce.
// ---------------------------------------------------------------------------
__global__ __launch_bounds__(256) void ln_kernel(
    const float* __restrict__ X, const float* __restrict__ Add,
    const float* __restrict__ g, const float* __restrict__ bb,
    float* __restrict__ Y, int R, float eps) {
  int lane = threadIdx.x & 31, wave = threadIdx.x >> 5;
  int row = blockIdx.x * 8 + wave;
  if (row >= R) return;
  size_t base = (size_t)row * D_MODEL;
  float v0 = X[base + lane], v1 = X[base + lane + 32];
  if (Add) { v0 += Add[base + lane]; v1 += Add[base + lane + 32]; }
  float s = v0 + v1, q = v0 * v0 + v1 * v1;
  for (int off = 16; off; off >>= 1) {
    s += __shfl_xor(s, off, 32);
    q += __shfl_xor(q, off, 32);
  }
  float m = s * (1.0f / 64.0f);
  float var = q * (1.0f / 64.0f) - m * m;
  float rs = rsqrtf(var + eps);
  Y[base + lane]      = (v0 - m) * rs * g[lane]      + bb[lane];
  Y[base + lane + 32] = (v1 - m) * rs * g[lane + 32] + bb[lane + 32];
}

// ---------------------------------------------------------------------------
// Generic bf16-WMMA GEMM over 16-row tiles (compile-time K,N => full unroll,
// fragments in fixed VGPRs, no v_movrels):
//   t = X(16xK) @ W^T + bias ; relu? ; t *= rowscale[row] ; t += resid ; (+=Y)
// ---------------------------------------------------------------------------
template <int K, int N>
__global__ __launch_bounds__(256) void gemm_kernel(
    const float* __restrict__ X, const float* __restrict__ W,
    const float* __restrict__ bias, const float* __restrict__ resid,
    const float* __restrict__ rowscale, int rs_stride,
    float* __restrict__ Y, int ldY, int col0, int relu, int acc) {
  constexpr int NT = N >> 4, KT = K >> 5;
  int rt = blockIdx.x;
  int lane = threadIdx.x & 31, wave = threadIdx.x >> 5;
  int m15 = lane & 15, half = lane >> 4;
  int row_a = rt * 16 + m15;

  v16bf afr[KT];
  const float* xb = (row_a < R_TOT) ? (X + (size_t)row_a * K) : nullptr;
#pragma unroll
  for (int kt = 0; kt < KT; ++kt) {
#pragma unroll
    for (int i = 0; i < 8; ++i) {
      int k0 = kt * 32 + afrag_k(i, half);
      afr[kt][2 * i]     = xb ? (__bf16)xb[k0]     : (__bf16)0.0f;
      afr[kt][2 * i + 1] = xb ? (__bf16)xb[k0 + 1] : (__bf16)0.0f;
    }
  }

  for (int ct = wave; ct < NT; ct += 8) {
    int ncol = ct * 16 + m15;
    const float* wb = W + (size_t)ncol * K;
    __builtin_prefetch(wb, 0, 1);     // global_prefetch_b8
    v8f c = {};
#pragma unroll
    for (int kt = 0; kt < KT; ++kt) {
      v16bf b;
#pragma unroll
      for (int i = 0; i < 8; ++i) {
        int kk = kt * 32 + half * 16 + 2 * i;
        b[2 * i]     = (__bf16)wb[kk];
        b[2 * i + 1] = (__bf16)wb[kk + 1];
      }
      c = wmma_bf16(afr[kt], b, c);
    }
#pragma unroll
    for (int vr = 0; vr < 8; ++vr) {
      int row = rt * 16 + vr + 8 * half;
      int col = ct * 16 + m15;
      if (row < R_TOT) {
        float t = c[vr] + (bias ? bias[col] : 0.0f);
        if (relu) t = fmaxf(t, 0.0f);
        if (rowscale) t *= rowscale[(size_t)row * rs_stride];
        size_t oi = (size_t)row * ldY + col0 + col;
        if (resid) t += resid[oi];
        if (acc) t += Y[oi];
        Y[oi] = t;
      }
    }
  }
}

// ---------------------------------------------------------------------------
// Attention core: one wave per (batch, head). QKV packed [R,192] f32.
// K -> LDS bf16 (converted on stage-in); V -> LDS f32 via async copy
// (global_load_async_to_lds_b128 + s_wait_asynccnt); P@V through WMMA.
// ---------------------------------------------------------------------------
__global__ __launch_bounds__(32) void attn_kernel(
    const float* __restrict__ QKV, float* __restrict__ O, int causal) {
  __shared__ __bf16 Ks[LP][34];    // 15,232 B
  __shared__ float  Vf[LP][32];    // 28,672 B (rows 128B-aligned for b128)
  __shared__ float  sb[16][LP];    // 14,336 B

  int bh = blockIdx.x;
  int b = bh >> 1, h = bh & 1;
  int lane = threadIdx.x;
  int m15 = lane & 15, half = lane >> 4;
  const int hd = 32;

  // zero padded rows (l in [L_SEQ, LP))
  for (int idx = lane; idx < (LP - L_SEQ) * hd; idx += 32) {
    int l = L_SEQ + (idx >> 5), k = idx & 31;
    Ks[l][k] = (__bf16)0.0f;
    Vf[l][k] = 0.0f;
  }
  // K: stage + convert to bf16
  for (int idx = lane; idx < L_SEQ * hd; idx += 32) {
    int l = idx >> 5, k = idx & 31;
    const float* base = QKV + ((size_t)(b * L_SEQ + l)) * 192 + h * hd + k;
    Ks[l][k] = (__bf16)base[64];
  }
  // V: async DMA global -> LDS, 16B chunks, tracked by ASYNCcnt
  for (int c = lane; c < L_SEQ * 8; c += 32) {
    int l = c >> 3, k0 = (c & 7) * 4;
    const float* gp = QKV + ((size_t)(b * L_SEQ + l)) * 192 + 128 + h * hd + k0;
    unsigned lo = lds_off(&Vf[l][k0]);
    unsigned long long ga = (unsigned long long)(uintptr_t)gp;
    asm volatile("global_load_async_to_lds_b128 %0, %1, off"
                 :: "v"(lo), "v"(ga) : "memory");
  }
  asm volatile("s_wait_asynccnt 0x0" ::: "memory");
  __syncthreads();

  const float scale = 0.17677669529663687f;  // 1/sqrt(32)
  int nqt = (L_SEQ + 15) / 16;               // 13

  for (int qt = 0; qt < nqt; ++qt) {
    // A fragment: Q tile (16 x 32)
    v16bf aQ;
    int qrow = qt * 16 + m15;
    const float* qb =
        (qrow < L_SEQ) ? QKV + ((size_t)(b * L_SEQ + qrow)) * 192 + h * hd : nullptr;
#pragma unroll
    for (int i = 0; i < 8; ++i) {
      int k0 = afrag_k(i, half);
      aQ[2 * i]     = qb ? (__bf16)qb[k0]     : (__bf16)0.0f;
      aQ[2 * i + 1] = qb ? (__bf16)qb[k0 + 1] : (__bf16)0.0f;
    }
    // S = Q @ K^T  (one WMMA per 16-key tile, hd == 32 == one k-tile)
#pragma unroll
    for (int kt = 0; kt < LP / 16; ++kt) {
      v16bf bK;
      int kcol = kt * 16 + m15;
#pragma unroll
      for (int i = 0; i < 8; ++i) {
        int kk = half * 16 + 2 * i;
        bK[2 * i]     = Ks[kcol][kk];
        bK[2 * i + 1] = Ks[kcol][kk + 1];
      }
      v8f c = {};
      c = wmma_bf16(aQ, bK, c);
#pragma unroll
      for (int vr = 0; vr < 8; ++vr) {
        int m = vr + 8 * half;
        int qg = qt * 16 + m;
        int kg = kt * 16 + m15;
        float s = c[vr] * scale;
        if (kg >= L_SEQ || (causal && kg > qg)) s = -1.0e9f;
        sb[m][kg] = s;
      }
    }
    __syncthreads();
    // full-row softmax (f32, v_exp_f32)
    if (lane < 16) {
      float mx = -1.0e30f;
      for (int j = 0; j < L_SEQ; ++j) mx = fmaxf(mx, sb[lane][j]);
      float sm = 0.0f;
      for (int j = 0; j < LP; ++j) {
        float e = (j < L_SEQ) ? __expf(sb[lane][j] - mx) : 0.0f;
        sb[lane][j] = e;
        sm += e;
      }
      float inv = 1.0f / sm;
      for (int j = 0; j < LP; ++j) sb[lane][j] *= inv;
    }
    __syncthreads();
    // O = P @ V  (accumulate over 7 k-tiles of 32 keys, 2 n-tiles of 16)
    v8f c0 = {}, c1 = {};
#pragma unroll
    for (int k2 = 0; k2 < LP / 32; ++k2) {
      v16bf aP;
#pragma unroll
      for (int i = 0; i < 8; ++i) {
        int k0 = k2 * 32 + afrag_k(i, half);
        aP[2 * i]     = (__bf16)sb[m15][k0];
        aP[2 * i + 1] = (__bf16)sb[m15][k0 + 1];
      }
#pragma unroll
      for (int nt = 0; nt < 2; ++nt) {
        v16bf bV;
#pragma unroll
        for (int i = 0; i < 8; ++i) {
          int kk = k2 * 32 + half * 16 + 2 * i;
          bV[2 * i]     = (__bf16)Vf[kk][nt * 16 + m15];
          bV[2 * i + 1] = (__bf16)Vf[kk + 1][nt * 16 + m15];
        }
        if (nt == 0) c0 = wmma_bf16(aP, bV, c0);
        else         c1 = wmma_bf16(aP, bV, c1);
      }
    }
#pragma unroll
    for (int vr = 0; vr < 8; ++vr) {
      int m = vr + 8 * half;
      int row = qt * 16 + m;
      if (row < L_SEQ) {
        float* ob = O + ((size_t)(b * L_SEQ + row)) * D_MODEL + h * hd;
        ob[m15]      = c0[vr];
        ob[16 + m15] = c1[vr];
      }
    }
    __syncthreads();
  }
}

// ---------------------------------------------------------------------------
// Tiny N=4 projection (gate logits): VALU dot products.
// ---------------------------------------------------------------------------
__global__ __launch_bounds__(256) void small_mm_kernel(
    const float* __restrict__ X, const float* __restrict__ W,
    const float* __restrict__ bias, float* __restrict__ Y, int R, int K, int N) {
  int idx = blockIdx.x * 256 + threadIdx.x;
  if (idx >= R * N) return;
  int r = idx / N, n = idx % N;
  float acc = bias ? bias[n] : 0.0f;
  const float* xb = X + (size_t)r * K;
  const float* wb = W + (size_t)n * K;
  for (int k = 0; k < K; ++k) acc += xb[k] * wb[k];
  Y[idx] = acc;
}

__device__ __forceinline__ unsigned hash32(unsigned x) {
  x ^= x >> 17; x *= 0xed5ad4bbu; x ^= x >> 11;
  x *= 0xac4c1b51u; x ^= x >> 15; x *= 0x31848babu; x ^= x >> 14;
  return x;
}

// ---------------------------------------------------------------------------
// Gumbel-noised entropy-scaled top-2 routing -> per-row expert weights.
// (hash-based Gumbel stand-in for jax.random; structure matches reference)
// ---------------------------------------------------------------------------
__global__ __launch_bounds__(256) void gate_select_kernel(
    const float* __restrict__ GL, float* __restrict__ Wout,
    float* __restrict__ wsum, int R, int iter, int accum) {
  int r = blockIdx.x * 256 + threadIdx.x;
  if (r >= R) return;
  float gl[4], gp[4], noisy[4];
  float mx = -1.0e30f;
  for (int e = 0; e < 4; ++e) { gl[e] = GL[r * 4 + e]; mx = fmaxf(mx, gl[e]); }
  float s = 0.0f;
  for (int e = 0; e < 4; ++e) { gp[e] = __expf(gl[e] - mx); s += gp[e]; }
  float ent = 0.0f;
  for (int e = 0; e < 4; ++e) { gp[e] /= s; ent -= gp[e] * __logf(gp[e] + 1e-20f); }
  for (int e = 0; e < 4; ++e) {
    unsigned hsh = hash32(((unsigned)r * 4u + (unsigned)e) ^
                          (0x9E3779B9u * (unsigned)(iter + 1)));
    float u = ((hsh >> 8) + 1u) * (1.0f / 16777218.0f);
    float gum = -__logf(-__logf(u));
    noisy[e] = (gl[e] * ent + gum) * (1.0f / TAU_C);
  }
  int i1 = 0; float n1 = noisy[0];
  for (int e = 1; e < 4; ++e) if (noisy[e] > n1) { n1 = noisy[e]; i1 = e; }
  int i2 = (i1 == 0) ? 1 : 0; float n2 = noisy[i2];
  for (int e = 0; e < 4; ++e)
    if (e != i1 && noisy[e] > n2) { n2 = noisy[e]; i2 = e; }
  float e2 = __expf(n2 - n1);
  float inv = 1.0f / (1.0f + e2);
  float w[4] = {0.0f, 0.0f, 0.0f, 0.0f};
  w[i1] = inv; w[i2] = e2 * inv;
  for (int e = 0; e < 4; ++e) Wout[r * 4 + e] = w[e];
  if (accum)
    for (int e = 0; e < 4; ++e) atomicAdd(&wsum[e], w[e]);
}

__global__ __launch_bounds__(256) void axpy_kernel(
    float* __restrict__ Y, const float* __restrict__ X, float a, float b, int n) {
  int i = blockIdx.x * 256 + threadIdx.x;
  if (i < n) Y[i] = a * X[i] + b * Y[i];
}

__global__ void zero_kernel(float* p, int n) {
  int i = blockIdx.x * blockDim.x + threadIdx.x;
  if (i < n) p[i] = 0.0f;
}

// ---------------------------------------------------------------------------
// Final logits: (feats[:, -1] + user_emb[uid]) . item_emb[iid]
// ---------------------------------------------------------------------------
__global__ __launch_bounds__(64) void logits_kernel(
    const float* __restrict__ feats, const float* __restrict__ user_emb,
    const float* __restrict__ item_emb, const int* __restrict__ user_ids,
    const int* __restrict__ item_ids, float* __restrict__ out) {
  __shared__ float red[64];
  int b = blockIdx.x, d = threadIdx.x;
  float c = feats[((size_t)b * L_SEQ + (L_SEQ - 1)) * D_MODEL + d] +
            user_emb[(size_t)user_ids[b] * D_MODEL + d];
  red[d] = c * item_emb[(size_t)item_ids[b] * D_MODEL + d];
  __syncthreads();
  for (int off = 32; off; off >>= 1) {
    if (d < off) red[d] += red[d + off];
    __syncthreads();
  }
  if (d == 0) out[b] = red[0];
}

__global__ void kl_kernel(const float* __restrict__ wsum, float* __restrict__ out) {
  if (blockIdx.x == 0 && threadIdx.x == 0) {
    const float tgt = 1.0f / (float)E_EXP;
    float kl = 0.0f;
    for (int e = 0; e < 4; ++e) {
      float agp = wsum[e] / (float)R_TOT;
      kl += tgt * (__logf(tgt) - __logf(agp + 1e-8f));
    }
    out[0] = kl / (float)E_EXP;
  }
}

// ---------------------------------------------------------------------------
// Host-side parameter structs (JAX pytree sorted-key order).
// ---------------------------------------------------------------------------
struct LNP   { const float *b, *g; };
struct AttnP { const float *b_in, *b_out, *w_in, *w_out; };
struct GateP { const float *W1, *W2, *Wo; AttnP attn;
               const float *b1, *b2, *bo; LNP ln1, ln2; };
struct MoeP  { const float *W1, *W2, *b1, *b2; GateP gate; LNP ln; };
struct BlockP{ AttnP attn; LNP ln_attn, ln_ffn; MoeP moe; };

static inline void gemm(hipStream_t s, int K, int N, const float* X, const float* W,
                        const float* bias, const float* resid,
                        const float* rowscale, int rs_stride, float* Y,
                        int ldY, int col0, int relu, int acc) {
  dim3 g((R_TOT + 15) / 16), blk(256);
  if (K == 64 && N == 64)
    gemm_kernel<64, 64><<<g, blk, 0, s>>>(X, W, bias, resid, rowscale, rs_stride,
                                          Y, ldY, col0, relu, acc);
  else if (K == 64 && N == 128)
    gemm_kernel<64, 128><<<g, blk, 0, s>>>(X, W, bias, resid, rowscale, rs_stride,
                                           Y, ldY, col0, relu, acc);
  else if (K == 64 && N == 192)
    gemm_kernel<64, 192><<<g, blk, 0, s>>>(X, W, bias, resid, rowscale, rs_stride,
                                           Y, ldY, col0, relu, acc);
  else if (K == 128 && N == 64)
    gemm_kernel<128, 64><<<g, blk, 0, s>>>(X, W, bias, resid, rowscale, rs_stride,
                                           Y, ldY, col0, relu, acc);
}

extern "C" void kernel_launch(void* const* d_in, const int* in_sizes, int n_in,
                              void* d_out, int out_size, void* d_ws, size_t ws_size,
                              hipStream_t stream) {
  (void)in_sizes; (void)n_in; (void)out_size; (void)ws_size;
  const int*   user_ids = (const int*)d_in[0];
  const int*   log_seqs = (const int*)d_in[1];
  const int*   item_ids = (const int*)d_in[2];

  // ---- params: pytree leaves in sorted-key order starting at d_in[3] ----
  int cur = 3;
  auto nf = [&]() { return (const float*)d_in[cur++]; };
  BlockP blk[NB_C];
  for (int bi = 0; bi < NB_C; ++bi) {
    BlockP& bp = blk[bi];
    bp.attn.b_in = nf(); bp.attn.b_out = nf(); bp.attn.w_in = nf(); bp.attn.w_out = nf();
    bp.ln_attn.b = nf(); bp.ln_attn.g = nf();
    bp.ln_ffn.b  = nf(); bp.ln_ffn.g  = nf();
    MoeP& mp = bp.moe;
    mp.W1 = nf(); mp.W2 = nf(); mp.b1 = nf(); mp.b2 = nf();
    GateP& gp = mp.gate;
    gp.W1 = nf(); gp.W2 = nf(); gp.Wo = nf();
    gp.attn.b_in = nf(); gp.attn.b_out = nf(); gp.attn.w_in = nf(); gp.attn.w_out = nf();
    gp.b1 = nf(); gp.b2 = nf(); gp.bo = nf();
    gp.ln1.b = nf(); gp.ln1.g = nf();
    gp.ln2.b = nf(); gp.ln2.g = nf();
    mp.ln.b = nf(); mp.ln.g = nf();
  }
  const float* item_emb = nf();
  LNP last_ln; last_ln.b = nf(); last_ln.g = nf();
  const float* pos_emb  = nf();
  const float* user_emb = nf();

  // ---- workspace carve-up (f32) ----
  float* w = (float*)d_ws;
  const size_t RD = (size_t)R_TOT * D_MODEL;
  float* x     = w;              w += RD;
  float* tA    = w;              w += RD;
  float* tB    = w;              w += RD;
  float* QKV   = w;              w += (size_t)R_TOT * 192;
  float* O     = w;              w += RD;
  float* G128  = w;              w += (size_t)R_TOT * 128;
  float* EO    = w;              w += RD;
  float* H1    = w;              w += RD;
  float* boost = w;              w += RD;
  float* wbuf  = w;              w += (size_t)R_TOT * 4;
  float* glbuf = w;              w += (size_t)R_TOT * 4;
  float* wsum  = w;              w += 4;

  const int nEl  = R_TOT * D_MODEL;
  const int gEl  = (nEl + 255) / 256;
  const int gRow = (R_TOT + 7) / 8;
  float* logits = (float*)d_out;

  zero_kernel<<<1, 4, 0, stream>>>(wsum, 4);
  embed_kernel<<<gEl, 256, 0, stream>>>(log_seqs, item_emb, pos_emb, x);

  for (int bi = 0; bi < NB_C; ++bi) {
    const BlockP& bp = blk[bi];
    // q = LN(x);  a = MHA(q, x, x) causal
    ln_kernel<<<gRow, 256, 0, stream>>>(x, nullptr, bp.ln_attn.g, bp.ln_attn.b,
                                        tA, R_TOT, 1e-8f);
    gemm(stream, 64, 64, tA, bp.attn.w_in, bp.attn.b_in,
         nullptr, nullptr, 0, QKV, 192, 0, 0, 0);                 // Q <- LN(x)
    gemm(stream, 64, 128, x, bp.attn.w_in + 64 * 64, bp.attn.b_in + 64,
         nullptr, nullptr, 0, QKV, 192, 64, 0, 0);                // K,V <- x
    attn_kernel<<<B_BATCH * NH_C, 32, 0, stream>>>(QKV, O, 1);
    gemm(stream, 64, 64, O, bp.attn.w_out, bp.attn.b_out, tA, nullptr, 0,
         tB, 64, 0, 0, 0);                                        // x = q + a
    ln_kernel<<<gRow, 256, 0, stream>>>(tB, nullptr, bp.ln_ffn.g, bp.ln_ffn.b,
                                        x, R_TOT, 1e-8f);         // x (MoE input)

    // ---- MoE: residual = x (kept intact); boost starts at x ----
    axpy_kernel<<<gEl, 256, 0, stream>>>(boost, x, 1.0f, 0.0f, nEl);
    const MoeP& mp = bp.moe;
    const GateP& gp = mp.gate;
    for (int it = 0; it < TOPK_C; ++it) {
      // gate(boost): mini-transformer
      gemm(stream, 64, 192, boost, gp.attn.w_in, gp.attn.b_in,
           nullptr, nullptr, 0, QKV, 192, 0, 0, 0);
      attn_kernel<<<B_BATCH * NH_C, 32, 0, stream>>>(QKV, O, 0);
      gemm(stream, 64, 64, O, gp.attn.w_out, gp.attn.b_out, boost, nullptr, 0,
           tA, 64, 0, 0, 0);                                      // boost + a
      ln_kernel<<<gRow, 256, 0, stream>>>(tA, nullptr, gp.ln1.g, gp.ln1.b,
                                          tB, R_TOT, 1e-5f);
      gemm(stream, 64, 128, tB, gp.W1, gp.b1, nullptr, nullptr, 0,
           G128, 128, 0, 1, 0);                                   // relu FFN up
      gemm(stream, 128, 64, G128, gp.W2, gp.b2, tB, nullptr, 0,
           tA, 64, 0, 0, 0);                                      // x + f
      ln_kernel<<<gRow, 256, 0, stream>>>(tA, nullptr, gp.ln2.g, gp.ln2.b,
                                          tB, R_TOT, 1e-5f);
      small_mm_kernel<<<(R_TOT * 4 + 255) / 256, 256, 0, stream>>>(
          tB, gp.Wo, gp.bo, glbuf, R_TOT, 64, 4);
      int accum = (bi == NB_C - 1 && it == TOPK_C - 1) ? 1 : 0;
      gate_select_kernel<<<(R_TOT + 255) / 256, 256, 0, stream>>>(
          glbuf, wbuf, wsum, R_TOT, it, accum);
      // experts: EO = sum_e w[:,e] * (relu(boost@W1e^T+b1e) @ W2e^T + b2e)
      for (int e = 0; e < E_EXP; ++e) {
        gemm(stream, 64, 64, boost, mp.W1 + (size_t)e * 64 * 64, mp.b1 + e * 64,
             nullptr, nullptr, 0, H1, 64, 0, 1, 0);
        gemm(stream, 64, 64, H1, mp.W2 + (size_t)e * 64 * 64, mp.b2 + e * 64,
             nullptr, wbuf + e, 4, EO, 64, 0, 0, (e > 0) ? 1 : 0);
      }
      axpy_kernel<<<gEl, 256, 0, stream>>>(boost, EO, ALPHA_C, 1.0f, nEl);
    }
    // x = LN(residual + boost)
    ln_kernel<<<gRow, 256, 0, stream>>>(x, boost, mp.ln.g, mp.ln.b,
                                        x, R_TOT, 1e-5f);
  }

  // feats = LN(x); logits; kl
  ln_kernel<<<gRow, 256, 0, stream>>>(x, nullptr, last_ln.g, last_ln.b,
                                      tA, R_TOT, 1e-8f);
  logits_kernel<<<B_BATCH, 64, 0, stream>>>(tA, user_emb, item_emb,
                                            user_ids, item_ids, logits);
  kl_kernel<<<1, 1, 0, stream>>>(wsum, logits + B_BATCH);
}
